// GIN_71725953843763
// MI455X (gfx1250) — matrix operand order repaired and checked by reading, not compiled, not run on hardware
//
#include <hip/hip_runtime.h>
#include <stdint.h>

typedef __attribute__((ext_vector_type(16))) _Float16 v16h;
typedef __attribute__((ext_vector_type(2)))  _Float16 v2h;
typedef __attribute__((ext_vector_type(8)))  float    v8f;

#define NNODES 50000
#define IN_C   64
#define HID_C  128
#define OUT_C  64

// ---------------------------------------------------------------------------
// Zero fp32 workspace (aggregation buffers must be zeroed every call)
// ---------------------------------------------------------------------------
__global__ void zero_f32(float* __restrict__ p, size_t n) {
    size_t i = (size_t)blockIdx.x * blockDim.x + threadIdx.x;
    size_t stride = (size_t)gridDim.x * blockDim.x;
    for (; i < n; i += stride) p[i] = 0.0f;
}

// ---------------------------------------------------------------------------
// Edge scatter-add:  agg[dst] += feat[src], C channels, float4 per thread
// edge_index is int64 per the reference (shape [2, E], row 0 = src, row 1 = dst)
// ---------------------------------------------------------------------------
template <int C>
__global__ void scatter_add(const float* __restrict__ feat,
                            const long long* __restrict__ ei,
                            float* __restrict__ agg, int nE) {
    constexpr int G = C / 4;
    long long idx = (long long)blockIdx.x * blockDim.x + threadIdx.x;
    if (idx >= (long long)nE * G) return;
    int e  = (int)(idx / G);
    int c0 = (int)(idx % G) * 4;
    long long s = ei[e];            // src node
    long long d = ei[(long long)nE + e];  // dst node
    const float4 v = *(const float4*)(feat + s * (long long)C + c0);
    float* o = agg + d * (long long)C + c0;
    atomicAdd(o + 0, v.x);
    atomicAdd(o + 1, v.y);
    atomicAdd(o + 2, v.z);
    atomicAdd(o + 3, v.w);
}

// ---------------------------------------------------------------------------
// Fused GIN MLP:  out = [relu?]( relu( (x+agg) @ Wa + ba ) @ Wb + bb )
// One wave (32 lanes) per 16-node tile; 2 waves per block.
// All matmuls via v_wmma_f32_16x16x32_f16 (fp32 accumulate).
//
// WMMA fragment layouts (CDNA5 ISA 7.12.2, wave32):
//   A 16x32 f16 : lane m = l&15, hi = l>>4; VGPR v holds K pair
//                 k = (v&4 ? 16:0) + hi*8 + (v&3)*2
//   B 32x16 f16 : lane n = l&15, hi = l>>4; VGPR v holds K pair k = (v&4 ? 16:0) + hi*8 + (v&3)*2 (same as A)
//   C/D 16x16 f32: lane n = l&15; VGPR v holds row m = v + (l>>4)*8
// Weights are stored in LDS pre-packed as (W[2k2][n], W[2k2+1][n]) v2h pairs
// so one B-fragment VGPR == one ds_load_b32.
// ---------------------------------------------------------------------------
template <int KIN, int KOUT, bool RELU_OUT>
__global__ __launch_bounds__(64) void gin_mlp(
    const float* __restrict__ xin, const float* __restrict__ agg,
    const float* __restrict__ Wa,  const float* __restrict__ ba,
    const float* __restrict__ Wb,  const float* __restrict__ bb,
    float* __restrict__ out, int nTiles)
{
    constexpr int KMID  = 128;
    constexpr int WAVES = 2;

    __shared__ v2h   s_wa[(KIN / 2) * KMID];   // packed f16 Wa
    __shared__ v2h   s_wb[(KMID / 2) * KOUT];  // packed f16 Wb
    __shared__ float s_ba[KMID];
    __shared__ float s_bb[KOUT];
    __shared__ v2h   s_stage[WAVES][16 * (KMID / 2)];  // 16 x 128 f16 per wave

    const int tid = threadIdx.x;

    // --- cooperative weight/bias load + fp32 -> packed f16 conversion ---
    for (int i = tid; i < (KIN / 2) * KMID; i += 64) {
        int k2 = i / KMID, n = i % KMID;
        v2h p = { (_Float16)Wa[(2 * k2) * KMID + n],
                  (_Float16)Wa[(2 * k2 + 1) * KMID + n] };
        s_wa[i] = p;
    }
    for (int i = tid; i < (KMID / 2) * KOUT; i += 64) {
        int k2 = i / KOUT, n = i % KOUT;
        v2h p = { (_Float16)Wb[(2 * k2) * KOUT + n],
                  (_Float16)Wb[(2 * k2 + 1) * KOUT + n] };
        s_wb[i] = p;
    }
    for (int i = tid; i < KMID; i += 64) s_ba[i] = ba[i];
    for (int i = tid; i < KOUT; i += 64) s_bb[i] = bb[i];
    __syncthreads();

    const int wave = tid >> 5;
    const int lane = tid & 31;
    const int tile = blockIdx.x * WAVES + wave;
    if (tile >= nTiles) return;   // whole-wave uniform exit (EXEC stays all-1s)

    const int m  = lane & 15;   // A-frag row / C-frag column id
    const int hi = lane >> 4;
    const long long row = (long long)tile * 16 + m;

    // --- stage 1 A fragments: t = x + agg, converted to f16 in A layout ---
    v16h a1[KIN / 32];
    {
        const float* xr = xin + row * KIN;
        const float* ar = agg + row * KIN;
#pragma unroll
        for (int c = 0; c < KIN / 32; ++c) {
#pragma unroll
            for (int v = 0; v < 8; ++v) {
                int k = c * 32 + ((v & 4) ? 16 : 0) + hi * 8 + (v & 3) * 2;
                float2 xv = *(const float2*)(xr + k);
                float2 av = *(const float2*)(ar + k);
                a1[c][2 * v]     = (_Float16)(xv.x + av.x);
                a1[c][2 * v + 1] = (_Float16)(xv.y + av.y);
            }
        }
    }

    v2h* stg = &s_stage[wave][0];  // per-wave 16 x (KMID/2) v2h stage buffer

    // --- stage 1: h = relu( t @ Wa + ba ), written to LDS stage ---
#pragma unroll
    for (int j = 0; j < KMID / 16; ++j) {
        v8f acc = {};
#pragma unroll
        for (int c = 0; c < KIN / 32; ++c) {
            v16h b;
#pragma unroll
            for (int v = 0; v < 8; ++v) {
                v2h p = s_wa[(c * 16 + ((v & 4) ? 8 : 0) + hi * 4 + (v & 3)) * KMID + j * 16 + m];
                b[2 * v]     = p.x;
                b[2 * v + 1] = p.y;
            }
            acc = __builtin_amdgcn_wmma_f32_16x16x32_f16(
                false, a1[c], false, b, (short)0, acc, false, false);
        }
#pragma unroll
        for (int v = 0; v < 8; ++v) {
            float val = acc[v] + s_ba[j * 16 + m];
            val = val > 0.0f ? val : 0.0f;
            int m2 = v + hi * 8;   // C-layout row
            ((_Float16*)stg)[m2 * KMID + j * 16 + m] = (_Float16)val;
        }
    }

    // same-wave LDS write->read ordering
    asm volatile("s_wait_dscnt 0" ::: "memory");

    // --- stage 2 A fragments from LDS stage (redistribute C -> A layout) ---
    v16h a2[KMID / 32];
#pragma unroll
    for (int c = 0; c < KMID / 32; ++c) {
#pragma unroll
        for (int v = 0; v < 8; ++v) {
            int k = c * 32 + ((v & 4) ? 16 : 0) + hi * 8 + (v & 3) * 2;
            v2h p = stg[m * (KMID / 2) + (k >> 1)];
            a2[c][2 * v]     = p.x;
            a2[c][2 * v + 1] = p.y;
        }
    }

    // --- stage 2: out = [relu?]( h @ Wb + bb ), streamed to global ---
#pragma unroll
    for (int j = 0; j < KOUT / 16; ++j) {
        v8f acc = {};
#pragma unroll
        for (int c = 0; c < KMID / 32; ++c) {
            v16h b;
#pragma unroll
            for (int v = 0; v < 8; ++v) {
                v2h p = s_wb[(c * 16 + ((v & 4) ? 8 : 0) + hi * 4 + (v & 3)) * KOUT + j * 16 + m];
                b[2 * v]     = p.x;
                b[2 * v + 1] = p.y;
            }
            acc = __builtin_amdgcn_wmma_f32_16x16x32_f16(
                false, a2[c], false, b, (short)0, acc, false, false);
        }
#pragma unroll
        for (int v = 0; v < 8; ++v) {
            float val = acc[v] + s_bb[j * 16 + m];
            if (RELU_OUT) val = val > 0.0f ? val : 0.0f;
            int m2 = v + hi * 8;
            out[((long long)tile * 16 + m2) * KOUT + j * 16 + m] = val;
        }
    }
}

// ---------------------------------------------------------------------------
// kernel_launch: zero -> scatter1 -> mlp1 -> scatter2 -> mlp2
// ---------------------------------------------------------------------------
extern "C" void kernel_launch(void* const* d_in, const int* in_sizes, int n_in,
                              void* d_out, int out_size, void* d_ws, size_t ws_size,
                              hipStream_t stream) {
    const float*      x  = (const float*)d_in[0];
    const long long*  ei = (const long long*)d_in[1];   // int64 [2, E]
    const float* W1 = (const float*)d_in[2];
    const float* b1 = (const float*)d_in[3];
    const float* W2 = (const float*)d_in[4];
    const float* b2 = (const float*)d_in[5];
    const float* W3 = (const float*)d_in[6];
    const float* b3 = (const float*)d_in[7];
    const float* W4 = (const float*)d_in[8];
    const float* b4 = (const float*)d_in[9];
    float* out = (float*)d_out;

    const int nE = in_sizes[1] / 2;
    const int nN = in_sizes[0] / IN_C;
    const int nTiles = (nN + 15) / 16;        // 3125 (N divisible by 16)

    float* ws   = (float*)d_ws;
    float* agg1 = ws;                                   // N * 64
    float* agg2 = agg1 + (size_t)nN * IN_C;             // N * 128
    float* hbuf = agg2 + (size_t)nN * HID_C;            // N * 128

    // zero agg1 + agg2 (contiguous region of N*(64+128) floats)
    {
        size_t nz = (size_t)nN * (IN_C + HID_C);
        int blocks = (int)((nz + 255) / 256);
        if (blocks > 65535) blocks = 65535;
        zero_f32<<<blocks, 256, 0, stream>>>(ws, nz);
    }

    // layer 1 scatter:  agg1[dst] += x[src]   (64 channels)
    {
        long long work = (long long)nE * (IN_C / 4);
        int blocks = (int)((work + 255) / 256);
        scatter_add<IN_C><<<blocks, 256, 0, stream>>>(x, ei, agg1, nE);
    }

    // GIN conv 1 + outer relu:  hbuf = relu( relu((x+agg1)W1+b1) W2 + b2 )
    gin_mlp<IN_C, HID_C, true><<<(nTiles + 1) / 2, 64, 0, stream>>>(
        x, agg1, W1, b1, W2, b2, hbuf, nTiles);

    // layer 2 scatter:  agg2[dst] += hbuf[src]   (128 channels)
    {
        long long work = (long long)nE * (HID_C / 4);
        int blocks = (int)((work + 255) / 256);
        scatter_add<HID_C><<<blocks, 256, 0, stream>>>(hbuf, ei, agg2, nE);
    }

    // GIN conv 2 (no outer relu):  out = relu((hbuf+agg2)W3+b3) W4 + b4
    gin_mlp<HID_C, OUT_C, false><<<(nTiles + 1) / 2, 64, 0, stream>>>(
        hbuf, agg2, W3, b3, W4, b4, out, nTiles);
}
